// MiMoV2MoE_7292854468630
// MI455X (gfx1250) — compile-verified, matrix operand order checked by reading, not run on hardware
//
#include <hip/hip_runtime.h>
#include <hip/hip_bf16.h>
#include <stdint.h>

#define NUM_EXPERTS 64
#define TOP_K 8
#define N_GROUP 8
#define TOPK_GROUP 4
#define HIDDEN 1024
#define INTER 512
#define TOKENS 2048
#define MTILE 32                 // tokens per block (2 WMMA M-tiles)

typedef __attribute__((ext_vector_type(16))) __bf16 v16bf;
typedef __attribute__((ext_vector_type(8)))  float  v8f;
typedef __attribute__((ext_vector_type(4)))  float  v4f;
typedef __attribute__((ext_vector_type(8)))  unsigned int v8u;

union BPack {
    v8u   u;
    v16bf v;
};

// round-to-nearest-even fp32 -> bf16, two at a time, packed into one dword
__device__ __forceinline__ unsigned pack2_bf16(float a, float b) {
    unsigned ua = __builtin_bit_cast(unsigned, a);
    unsigned ub = __builtin_bit_cast(unsigned, b);
    ua += 0x7FFFu + ((ua >> 16) & 1u);
    ub += 0x7FFFu + ((ub >> 16) & 1u);
    return (ua >> 16) | (ub & 0xFFFF0000u);
}

__device__ __forceinline__ v8f wmma_bf16(const BPack& a, const BPack& b, v8f c) {
    return __builtin_amdgcn_wmma_f32_16x16x32_bf16(
        /*neg_a=*/false, a.v, /*neg_b=*/false, b.v,
        /*c_mod=*/(short)0, c, /*reuse_a=*/false, /*reuse_b=*/false);
}

// ---- CDNA5 async global->LDS copy (16B per lane), tracked by ASYNCcnt ------
__device__ __forceinline__ void async_ld_b128(unsigned lds_byte_off, const void* g) {
    asm volatile("global_load_async_to_lds_b128 %0, %1, off"
                 :: "v"(lds_byte_off),
                    "v"((unsigned long long)(uintptr_t)g)
                 : "memory");
}
__device__ __forceinline__ void wait_async0() {
    asm volatile("s_wait_asynccnt 0" ::: "memory");
}

// ---------------------------------------------------------------- zero pass
__global__ void k_zero(float* __restrict__ y, int* __restrict__ counts) {
    int i = blockIdx.x * blockDim.x + threadIdx.x;
    if (i < TOKENS * HIDDEN) y[i] = 0.0f;
    if (i < NUM_EXPERTS) counts[i] = 0;
}

// ---------------------------------------------------------------- router
__global__ void k_router(const float* __restrict__ x,
                         const float* __restrict__ gate_w,
                         float* __restrict__ scores) {
    int t = blockIdx.x;
    int e = threadIdx.x;
    const v4f* xr = (const v4f*)(x + (size_t)t * HIDDEN);
    const v4f* gr = (const v4f*)(gate_w + (size_t)e * HIDDEN);
    float acc = 0.0f;
#pragma unroll 4
    for (int i = 0; i < HIDDEN / 4; ++i) {
        v4f a = xr[i], b = gr[i];
        acc += a.x * b.x + a.y * b.y + a.z * b.z + a.w * b.w;
    }
    scores[(size_t)t * NUM_EXPERTS + e] = 1.0f / (1.0f + __expf(-acc));
}

// ------------------------------------------------ fp32 -> packed bf16 (x4/thread)
__global__ void k_cvt(const float* __restrict__ in, uint2* __restrict__ out, int nquads) {
    int i = blockIdx.x * blockDim.x + threadIdx.x;
    if (i < nquads) {
        float4 v = ((const float4*)in)[i];
        out[i] = make_uint2(pack2_bf16(v.x, v.y), pack2_bf16(v.z, v.w));
    }
}

// ---------------------------------------------------------------- grouped top-k
__global__ void k_topk(const float* __restrict__ scores,
                       const float* __restrict__ bias,
                       int* __restrict__ counts,
                       int* __restrict__ etok,
                       float* __restrict__ ewt) {
    int t = blockIdx.x * blockDim.x + threadIdx.x;
    if (t >= TOKENS) return;
    const float NEG = -1e30f;

    float s[NUM_EXPERTS], sel[NUM_EXPERTS];
    for (int e = 0; e < NUM_EXPERTS; ++e) {
        s[e] = scores[(size_t)t * NUM_EXPERTS + e];
        sel[e] = s[e] + bias[e];
    }
    float gs[N_GROUP];
    for (int g = 0; g < N_GROUP; ++g) {
        float m1 = NEG, m2 = NEG;
        for (int j = 0; j < NUM_EXPERTS / N_GROUP; ++j) {
            float v = sel[g * (NUM_EXPERTS / N_GROUP) + j];
            if (v > m1) { m2 = m1; m1 = v; }
            else if (v > m2) { m2 = v; }
        }
        gs[g] = m1 + m2;
    }
    bool gmask[N_GROUP];
    for (int g = 0; g < N_GROUP; ++g) gmask[g] = false;
    for (int r = 0; r < TOPK_GROUP; ++r) {
        int bg = 0; float bv = NEG;
        for (int g = 0; g < N_GROUP; ++g)
            if (gs[g] > bv) { bv = gs[g]; bg = g; }
        gmask[bg] = true;
        gs[bg] = NEG;
    }
    for (int e = 0; e < NUM_EXPERTS; ++e)
        if (!gmask[e >> 3]) sel[e] = NEG;
    int   ids[TOP_K];
    float w[TOP_K];
    float wsum = 0.0f;
    for (int r = 0; r < TOP_K; ++r) {
        int bi = 0; float bv = NEG;
        for (int e = 0; e < NUM_EXPERTS; ++e)
            if (sel[e] > bv) { bv = sel[e]; bi = e; }
        ids[r] = bi;
        w[r] = s[bi];            // weight uses raw (un-biased) sigmoid score
        wsum += w[r];
        sel[bi] = NEG;
    }
    float inv = 1.0f / wsum;
    for (int r = 0; r < TOP_K; ++r) {
        int e = ids[r];
        int p = atomicAdd(&counts[e], 1);
        etok[(size_t)e * TOKENS + p] = t;
        ewt[(size_t)e * TOKENS + p] = w[r] * inv;
    }
}

// -------- B-fragment loader: bf16 direct (preferred) or fp32 + convert ------
template <bool BF16W>
__device__ __forceinline__ BPack load_b_frag(const void* wbase, int col, int kb, int ldk) {
    BPack bf;
    if constexpr (BF16W) {
        const unsigned short* p = (const unsigned short*)wbase + (size_t)col * ldk + kb;
        __builtin_prefetch((const char*)p + 256, 0, 0);    // 4 K-slabs ahead
        const uint4* q = (const uint4*)p;
        uint4 lo = q[0], hi = q[1];
        bf.u[0] = lo.x; bf.u[1] = lo.y; bf.u[2] = lo.z; bf.u[3] = lo.w;
        bf.u[4] = hi.x; bf.u[5] = hi.y; bf.u[6] = hi.z; bf.u[7] = hi.w;
    } else {
        const float* bp = (const float*)wbase + (size_t)col * ldk + kb;
        __builtin_prefetch(bp + 128, 0, 0);
        v4f b0 = *(const v4f*)(bp);
        v4f b1 = *(const v4f*)(bp + 4);
        v4f b2 = *(const v4f*)(bp + 8);
        v4f b3 = *(const v4f*)(bp + 12);
        bf.u[0] = pack2_bf16(b0.x, b0.y); bf.u[1] = pack2_bf16(b0.z, b0.w);
        bf.u[2] = pack2_bf16(b1.x, b1.y); bf.u[3] = pack2_bf16(b1.z, b1.w);
        bf.u[4] = pack2_bf16(b2.x, b2.y); bf.u[5] = pack2_bf16(b2.z, b2.w);
        bf.u[6] = pack2_bf16(b3.x, b3.y); bf.u[7] = pack2_bf16(b3.z, b3.w);
    }
    return bf;
}

__device__ __forceinline__ BPack load_a_frag_lds(const unsigned short* p) {
    BPack af;
    const uint4* ap = (const uint4*)p;
    uint4 lo = ap[0], hi = ap[1];
    af.u[0] = lo.x; af.u[1] = lo.y; af.u[2] = lo.z; af.u[3] = lo.w;
    af.u[4] = hi.x; af.u[5] = hi.y; af.u[6] = hi.z; af.u[7] = hi.w;
    return af;
}

// ---------------------------------------------------------------- fused expert FFN
// block = 256 threads = 8 wave32s; grid = (max M-tiles, experts); M-tile = 32 tokens
// LDS layout (dynamic, ~227 KB):
//   [0)       32 x 1040 ushort  bf16 gathered A (x rows), padded stride
//   [66560)   32 x 1024 float   gu intermediate
//   [197632)  32 x 528  ushort  bf16 h (SwiGLU output), padded stride
//   [231424)  32 int            token ids
//   [231552)  32 float          routing weights
#define LDS_A_OFF  0
#define LDS_GU_OFF 66560
#define LDS_H_OFF  197632
#define LDS_TK_OFF 231424
#define LDS_RW_OFF 231552
#define LDS_BYTES  231680

template <bool BF16W>
__global__ __launch_bounds__(256) void k_moe(
    const unsigned short* __restrict__ xb,
    const void* __restrict__ w13x,    // bf16 (BF16W) or fp32 weights, [E][2I][H]
    const void* __restrict__ w2x,     // bf16 (BF16W) or fp32 weights, [E][H][I]
    const int* __restrict__ counts,
    const int* __restrict__ etok,
    const float* __restrict__ ewt,
    float* __restrict__ y) {
    const int e    = blockIdx.y;
    const int tile = blockIdx.x;
    const int cnt  = counts[e];
    if (tile * MTILE >= cnt) return;

    extern __shared__ char smem[];
    unsigned short* lds_a  = (unsigned short*)(smem + LDS_A_OFF);
    float*          lds_gu = (float*)(smem + LDS_GU_OFF);
    unsigned short* lds_h  = (unsigned short*)(smem + LDS_H_OFF);
    int*            lds_tk = (int*)(smem + LDS_TK_OFF);
    float*          lds_rw = (float*)(smem + LDS_RW_OFF);

    const int tid  = threadIdx.x;
    const int lane = tid & 31;       // wave32
    const int wv   = tid >> 5;       // 8 waves

    if (tid < MTILE) {
        int idx = tile * MTILE + tid;
        int tk = -1; float rw = 0.0f;
        if (idx < cnt) {
            tk = etok[(size_t)e * TOKENS + idx];
            rw = ewt[(size_t)e * TOKENS + idx];
        }
        lds_tk[tid] = tk;
        lds_rw[tid] = rw;
    }
    __syncthreads();

    // ---- async-gather 32 token rows (bf16) into LDS: 256 B per thread ------
    {
        int row = tid >> 3;     // 0..31
        int sub = tid & 7;      // 0..7, 256 bytes each
        int tk = lds_tk[row];
        if (tk < 0) tk = 0;     // dummy row: its outputs are dropped in epilogue
        unsigned lbase = (unsigned)(uintptr_t)((char*)lds_a + row * 2080 + sub * 256);
        const char* g = (const char*)xb + (size_t)tk * (HIDDEN * 2) + sub * 256;
#pragma unroll
        for (int i = 0; i < 16; ++i) async_ld_b128(lbase + i * 16, g + i * 16);
    }
    wait_async0();
    __syncthreads();

    const int arow  = lane & 15;               // A-matrix row within M-tile
    const int khalf = (lane >> 4) * 16;        // K half (0 or 16) for this lane
    const int ncb   = wv * 128 + (lane & 15);  // base output column for this lane

    // ================= GEMM1: gu[32, 1024] = A[32,1024] @ w13_e^T ============
    const char* w13e = (const char*)w13x +
                       (size_t)e * (2 * INTER) * HIDDEN * (BF16W ? 2 : 4);
    v8f acc0[8] = {};
    v8f acc1[8] = {};
    for (int kt = 0; kt < HIDDEN / 32; ++kt) {
        BPack af0 = load_a_frag_lds(lds_a + arow * 1040 + kt * 32 + khalf);
        BPack af1 = load_a_frag_lds(lds_a + (arow + 16) * 1040 + kt * 32 + khalf);
        const int kb = kt * 32 + khalf;
        BPack bf = load_b_frag<BF16W>(w13e, ncb, kb, HIDDEN);
#pragma unroll
        for (int nt = 0; nt < 8; ++nt) {
            BPack bfn;
            if (nt < 7) bfn = load_b_frag<BF16W>(w13e, ncb + (nt + 1) * 16, kb, HIDDEN);
            acc0[nt] = wmma_bf16(af0, bf, acc0[nt]);
            acc1[nt] = wmma_bf16(af1, bf, acc1[nt]);
            if (nt < 7) bf = bfn;
        }
    }
    // spill gu to LDS (C/D layout: vgpr j, lanes 0-15 -> M=j, lanes 16-31 -> M=j+8)
#pragma unroll
    for (int nt = 0; nt < 8; ++nt) {
        int col = wv * 128 + nt * 16 + (lane & 15);
#pragma unroll
        for (int j = 0; j < 8; ++j) {
            int row = j + ((lane >> 4) << 3);
            lds_gu[row * 1024 + col]        = acc0[nt][j];
            lds_gu[(row + 16) * 1024 + col] = acc1[nt][j];
        }
    }
    __syncthreads();

    // ================= SwiGLU: h = silu(g) * u, bf16 into LDS ================
#pragma unroll 4
    for (int i = 0; i < 32; ++i) {
        int p   = tid * 32 + i;       // 8192 bf16-pairs total
        int row = p >> 8;
        int col = (p & 255) * 2;
        float g0 = lds_gu[row * 1024 + col];
        float g1 = lds_gu[row * 1024 + col + 1];
        float u0 = lds_gu[row * 1024 + 512 + col];
        float u1 = lds_gu[row * 1024 + 512 + col + 1];
        float h0 = (g0 / (1.0f + __expf(-g0))) * u0;
        float h1 = (g1 / (1.0f + __expf(-g1))) * u1;
        ((unsigned*)(lds_h + row * 528))[col >> 1] = pack2_bf16(h0, h1);
    }
    __syncthreads();

    // ================= GEMM2: out[32,1024] = h[32,512] @ w2_e^T ==============
    const char* w2e = (const char*)w2x +
                      (size_t)e * HIDDEN * INTER * (BF16W ? 2 : 4);
    v8f bcc0[8] = {};
    v8f bcc1[8] = {};
    for (int kt = 0; kt < INTER / 32; ++kt) {
        BPack af0 = load_a_frag_lds(lds_h + arow * 528 + kt * 32 + khalf);
        BPack af1 = load_a_frag_lds(lds_h + (arow + 16) * 528 + kt * 32 + khalf);
        const int kb = kt * 32 + khalf;
        BPack bf = load_b_frag<BF16W>(w2e, ncb, kb, INTER);
#pragma unroll
        for (int nt = 0; nt < 8; ++nt) {
            BPack bfn;
            if (nt < 7) bfn = load_b_frag<BF16W>(w2e, ncb + (nt + 1) * 16, kb, INTER);
            bcc0[nt] = wmma_bf16(af0, bf, bcc0[nt]);
            bcc1[nt] = wmma_bf16(af1, bf, bcc1[nt]);
            if (nt < 7) bf = bfn;
        }
    }

    // ================= epilogue: weighted scatter-accumulate into y ==========
    int   tkv0[8], tkv1[8];
    float rwv0[8], rwv1[8];
#pragma unroll
    for (int j = 0; j < 8; ++j) {
        int row = j + ((lane >> 4) << 3);
        tkv0[j] = lds_tk[row];       rwv0[j] = lds_rw[row];
        tkv1[j] = lds_tk[row + 16];  rwv1[j] = lds_rw[row + 16];
    }
#pragma unroll
    for (int nt = 0; nt < 8; ++nt) {
        int col = wv * 128 + nt * 16 + (lane & 15);
#pragma unroll
        for (int j = 0; j < 8; ++j) {
            if (tkv0[j] >= 0)
                atomicAdd(&y[(size_t)tkv0[j] * HIDDEN + col], rwv0[j] * bcc0[nt][j]);
            if (tkv1[j] >= 0)
                atomicAdd(&y[(size_t)tkv1[j] * HIDDEN + col], rwv1[j] * bcc1[nt][j]);
        }
    }
}

// ---------------------------------------------------------------- launcher
extern "C" void kernel_launch(void* const* d_in, const int* in_sizes, int n_in,
                              void* d_out, int out_size, void* d_ws, size_t ws_size,
                              hipStream_t stream) {
    const float* x      = (const float*)d_in[0];
    const float* gate_w = (const float*)d_in[1];
    const float* bias   = (const float*)d_in[2];
    const float* w13    = (const float*)d_in[3];
    const float* w2     = (const float*)d_in[4];
    float* y = (float*)d_out;

    // workspace layout
    char* ws = (char*)d_ws;
    float*          scores = (float*)(ws);                        // 512 KB
    int*            counts = (int*)(ws + 512 * 1024);             // 256 B (1 KB pad)
    int*            etok   = (int*)(ws + 513 * 1024);             // 512 KB
    float*          ewt    = (float*)(ws + 1025 * 1024);          // 512 KB
    unsigned short* xb     = (unsigned short*)(ws + 1537 * 1024); // 4 MB
    const size_t W13B_OFF   = 8ull << 20;                         // bf16 w13: 128 MiB
    const size_t W13B_BYTES = (size_t)NUM_EXPERTS * 2 * INTER * HIDDEN * 2;
    const size_t W2B_OFF    = W13B_OFF + W13B_BYTES;              // bf16 w2: 64 MiB
    const size_t W2B_BYTES  = (size_t)NUM_EXPERTS * HIDDEN * INTER * 2;
    unsigned short* w13b = (unsigned short*)(ws + W13B_OFF);
    unsigned short* w2b  = (unsigned short*)(ws + W2B_OFF);
    const bool use_bf16w = ws_size >= (W2B_OFF + W2B_BYTES);

    k_zero<<<(TOKENS * HIDDEN + 255) / 256, 256, 0, stream>>>(y, counts);
    k_router<<<TOKENS, NUM_EXPERTS, 0, stream>>>(x, gate_w, scores);
    {   // x -> bf16
        int nq = TOKENS * HIDDEN / 4;
        k_cvt<<<(nq + 255) / 256, 256, 0, stream>>>(x, (uint2*)xb, nq);
    }
    k_topk<<<TOKENS / 256, 256, 0, stream>>>(scores, bias, counts, etok, ewt);

    dim3 grid(TOKENS / MTILE, NUM_EXPERTS);
    if (use_bf16w) {
        int nq13 = NUM_EXPERTS * 2 * INTER * HIDDEN / 4;
        int nq2  = NUM_EXPERTS * HIDDEN * INTER / 4;
        k_cvt<<<(nq13 + 255) / 256, 256, 0, stream>>>(w13, (uint2*)w13b, nq13);
        k_cvt<<<(nq2 + 255) / 256, 256, 0, stream>>>(w2, (uint2*)w2b, nq2);
        k_moe<true><<<grid, 256, LDS_BYTES, stream>>>(xb, w13b, w2b,
                                                      counts, etok, ewt, y);
    } else {
        k_moe<false><<<grid, 256, LDS_BYTES, stream>>>(xb, w13, w2,
                                                       counts, etok, ewt, y);
    }
}
